// TransformerDecoder_39994735460722
// MI455X (gfx1250) — matrix operand order repaired and checked by reading, not compiled, not run on hardware
//
#include <hip/hip_runtime.h>

// ---------------------------------------------------------------------------
// Types for CDNA5 WMMA (gfx1250, wave32)
// ---------------------------------------------------------------------------
typedef __bf16  bf16_t;
typedef bf16_t  v16bf __attribute__((ext_vector_type(16)));
typedef float   v8f   __attribute__((ext_vector_type(8)));

// Model constants
#define VOCAB 32000
#define DMODEL 1024
#define NHEAD 16
#define DHEAD 64
#define FFN 4096
#define NLAYER 6
#define BATCH 2
#define SEQ 1024
#define MTOK (BATCH*SEQ)

// GEMM tiling
#define BM 128
#define BN 128
#define BK 32
#define LDSS 40   // padded LDS row stride (elements); 80B => 16B-aligned rows

static __device__ __forceinline__ unsigned short f2bf(float f) {
    unsigned u = __float_as_uint(f);
    return (unsigned short)((u + 0x7FFFu + ((u >> 16) & 1u)) >> 16);
}

// ---------------------------------------------------------------------------
// Embedding + sinusoidal positional encoding:  h = embed[x]*sqrt(D) + PE(s)
// ---------------------------------------------------------------------------
__global__ __launch_bounds__(256)
void embed_pe_kernel(const int* __restrict__ x, const float* __restrict__ embed,
                     float* __restrict__ h)
{
    const int row = blockIdx.x;            // 0..MTOK-1 (b*SEQ + s)
    const int s   = row % SEQ;
    const int tok = x[row];
    const float lg = logf(10000.0f) / (float)DMODEL;
    for (int c = threadIdx.x; c < DMODEL; c += 256) {
        float e = embed[(size_t)tok * DMODEL + c] * 32.0f;   // sqrt(1024)=32
        int   i2 = c & ~1;
        float freq = expf(-(float)i2 * lg);
        float arg  = (float)s * freq;
        float pe   = (c & 1) ? cosf(arg) : sinf(arg);
        h[(size_t)row * DMODEL + c] = e + pe;
    }
}

// ---------------------------------------------------------------------------
// f32 -> bf16 (RNE), elementwise
// ---------------------------------------------------------------------------
__global__ __launch_bounds__(256)
void conv_bf16_kernel(const float* __restrict__ in, unsigned short* __restrict__ out,
                      int n)
{
    int i = blockIdx.x * 256 + threadIdx.x;
    if (i < n) out[i] = f2bf(in[i]);
}

// ---------------------------------------------------------------------------
// Transpose + convert: in f32 [K][N] row-major  ->  out bf16 [N][K] row-major
// ---------------------------------------------------------------------------
__global__ __launch_bounds__(256)
void transpose_conv_kernel(const float* __restrict__ in, unsigned short* __restrict__ out,
                           int K, int N)
{
    __shared__ unsigned short tile[32][33];
    const int nb = blockIdx.x * 32;
    const int kb = blockIdx.y * 32;
    const int tx = threadIdx.x;
    const int ty = threadIdx.y;
    for (int j = ty; j < 32; j += 8)
        tile[j][tx] = f2bf(in[(size_t)(kb + j) * N + (nb + tx)]);
    __syncthreads();
    for (int j = ty; j < 32; j += 8)
        out[(size_t)(nb + j) * K + (kb + tx)] = tile[tx][j];
}

// ---------------------------------------------------------------------------
// bf16 WMMA GEMM:  C[M][N] = A[M][K] @ Bt[N][K]^T + bias
// mode: bit0 = ReLU, bit1 = bf16 output (else f32)
// Async double-buffered LDS staging (GLOBAL_LOAD_ASYNC_TO_LDS_B128, ASYNCcnt).
// Requires M%128==0, N%128==0, K%32==0 (true for all calls here).
// ---------------------------------------------------------------------------
__global__ __launch_bounds__(256)
void gemm_bf16_kernel(const unsigned short* __restrict__ A,
                      const unsigned short* __restrict__ Bt,
                      const float* __restrict__ bias,
                      void* __restrict__ Cv,
                      int M, int N, int K, int mode)
{
    __shared__ unsigned short As[2][BM * LDSS];
    __shared__ unsigned short Bs[2][BN * LDSS];

    const int t     = threadIdx.x;
    const int lane  = t & 31;
    const int wave  = t >> 5;
    const int wm    = wave >> 1;          // 0..3 -> M offset wm*32
    const int wn    = wave & 1;           // 0..1 -> N offset wn*64
    const int lh    = (lane < 16) ? 0 : 1;
    const int lmod  = lane & 15;
    const int m0    = blockIdx.y * BM;
    const int n0    = blockIdx.x * BN;

    const int lr = t >> 1;                // staging row 0..127
    const int lc = (t & 1) * 16;          // staging col 0 or 16

    const unsigned short* gA = A  + (size_t)(m0 + lr) * K + lc;
    const unsigned short* gB = Bt + (size_t)(n0 + lr) * K + lc;

    unsigned ldsA[2], ldsB[2];
    ldsA[0] = (unsigned)(size_t)&As[0][lr * LDSS + lc];
    ldsA[1] = (unsigned)(size_t)&As[1][lr * LDSS + lc];
    ldsB[0] = (unsigned)(size_t)&Bs[0][lr * LDSS + lc];
    ldsB[1] = (unsigned)(size_t)&Bs[1][lr * LDSS + lc];

    const int nk = K / BK;

#define ISSUE_TILE(buf, kidx)                                                 \
    do {                                                                      \
        const unsigned short* pa_ = gA + (kidx) * BK;                         \
        const unsigned short* pb_ = gB + (kidx) * BK;                         \
        asm volatile("global_load_async_to_lds_b128 %0, %1, off"              \
                     :: "v"(ldsA[buf]), "v"(pa_) : "memory");                 \
        asm volatile("global_load_async_to_lds_b128 %0, %1, off offset:16"    \
                     :: "v"(ldsA[buf]), "v"(pa_) : "memory");                 \
        asm volatile("global_load_async_to_lds_b128 %0, %1, off"              \
                     :: "v"(ldsB[buf]), "v"(pb_) : "memory");                 \
        asm volatile("global_load_async_to_lds_b128 %0, %1, off offset:16"    \
                     :: "v"(ldsB[buf]), "v"(pb_) : "memory");                 \
    } while (0)

    ISSUE_TILE(0, 0);
    if (nk > 1) ISSUE_TILE(1, 1);

    v8f acc[2][4] = {};

    for (int i = 0; i < nk; ++i) {
        const int cur = i & 1;
        if (i + 1 < nk) asm volatile("s_wait_asynccnt 0x4" ::: "memory");
        else            asm volatile("s_wait_asynccnt 0x0" ::: "memory");
        __syncthreads();                  // all waves' tile i landed

        v16bf bfrag[4];
        #pragma unroll
        for (int ni = 0; ni < 4; ++ni) {
            union { unsigned u[8]; v16bf v; } tb;
            const int nl = wn * 64 + ni * 16 + lmod;
            #pragma unroll
            for (int r = 0; r < 8; ++r)
                tb.u[r] = *(const unsigned*)&Bs[cur][nl * LDSS + ((r < 4) ? (lh * 8 + 2 * r) : (16 + lh * 8 + 2 * (r - 4)))];
            bfrag[ni] = tb.v;
        }
        #pragma unroll
        for (int mi = 0; mi < 2; ++mi) {
            union { unsigned u[8]; v16bf v; } ta;
            const int ml = wm * 32 + mi * 16 + lmod;
            #pragma unroll
            for (int r = 0; r < 8; ++r) {
                const int kk = (r < 4) ? (lh * 8 + 2 * r)
                                       : (16 + lh * 8 + 2 * (r - 4));
                ta.u[r] = *(const unsigned*)&As[cur][ml * LDSS + kk];
            }
            #pragma unroll
            for (int ni = 0; ni < 4; ++ni)
                acc[mi][ni] = __builtin_amdgcn_wmma_f32_16x16x32_bf16(
                    false, ta.v, false, bfrag[ni], (short)0, acc[mi][ni],
                    false, false);
        }

        __syncthreads();                  // all waves done reading buffer cur
        if (i + 2 < nk) ISSUE_TILE(cur, i + 2);
    }
#undef ISSUE_TILE

    // epilogue: bias (+ ReLU), f32 or bf16 store; dims tile-divisible.
    #pragma unroll
    for (int mi = 0; mi < 2; ++mi) {
        #pragma unroll
        for (int ni = 0; ni < 4; ++ni) {
            const int col = n0 + wn * 64 + ni * 16 + lmod;
            const float bc = bias[col];
            #pragma unroll
            for (int r = 0; r < 8; ++r) {
                const int row = m0 + wm * 32 + mi * 16 + r + lh * 8;
                float v = acc[mi][ni][r] + bc;
                if (mode & 1) v = fmaxf(v, 0.0f);
                if (mode & 2)
                    ((unsigned short*)Cv)[(size_t)row * N + col] = f2bf(v);
                else
                    ((float*)Cv)[(size_t)row * N + col] = v;
            }
        }
    }
}

// ---------------------------------------------------------------------------
// WMMA flash attention (causal, online softmax in C-fragment layout).
// Grid: (SEQ/128, BATCH*NHEAD), 256 threads = 8 waves.
// Each wave owns a 16-row q tile; 32 keys per iteration.
//   scores: A = q[16x64] (2 frags), B = k (LDS [key][dh], natural layout)
//   ctx   : A = P[16x32] (repacked via per-wave LDS scratch), B = vT [dh][key]
// Row softmax stats: row = (vgpr r, lane-half); reduce over 16 lanes with
// __shfl_xor masks 1/2/4/8 (stay inside the 16-lane group).
// q,k,v,ctx are bf16 [MTOK][DMODEL] with head-major column layout.
// ---------------------------------------------------------------------------
#define AQB 128   // queries per block
#define AKB 32    // keys per iteration
#define KSTR 72   // qs/ks LDS row stride (elements, 144B: 16B-aligned)
#define VSTR 34   // vT LDS row stride
#define PSTR 36   // P scratch row stride

__global__ __launch_bounds__(256)
void attn_wmma_kernel(const unsigned short* __restrict__ qB,
                      const unsigned short* __restrict__ kB,
                      const unsigned short* __restrict__ vB,
                      unsigned short* __restrict__ ctxB)
{
    __shared__ unsigned short qs[AQB * KSTR];     // 18.0 KB
    __shared__ unsigned short ks[AKB * KSTR];     //  4.5 KB
    __shared__ unsigned short vT[DHEAD * VSTR];   //  4.3 KB
    __shared__ unsigned short ps[8][16 * PSTR];   //  9.0 KB (per-wave scratch)

    const int bh   = blockIdx.y;
    const int b    = bh / NHEAD, h = bh % NHEAD;
    const int q0   = blockIdx.x * AQB;
    const int t    = threadIdx.x;
    const int lane = t & 31, w = t >> 5;
    const int lh   = lane >> 4, lmod = lane & 15;

    // ---- stage q tile once: 128x64 bf16, 4 x uint4 per thread ----
    #pragma unroll
    for (int it = 0; it < 4; ++it) {
        const int e   = (t + it * 256) * 8;
        const int row = e >> 6;
        const int c   = e & 63;
        const uint4* src = (const uint4*)(qB +
            ((size_t)(b * SEQ + q0 + row)) * DMODEL + h * DHEAD + c);
        *(uint4*)&qs[row * KSTR + c] = src[0];
    }
    __syncthreads();

    // per-wave q fragments (K=0..31, 32..63)
    v16bf qfrag[2];
    {
        const int qrow = w * 16 + lmod;
        #pragma unroll
        for (int f = 0; f < 2; ++f) {
            union { unsigned u[8]; v16bf v; } tq_;
            #pragma unroll
            for (int r = 0; r < 8; ++r) {
                const int kk = (r < 4) ? (lh * 8 + 2 * r)
                                       : (16 + lh * 8 + 2 * (r - 4));
                tq_.u[r] = *(const unsigned*)&qs[qrow * KSTR + f * 32 + kk];
            }
            qfrag[f] = tq_.v;
        }
    }

    v8f accC[4] = {};
    float m_run[8], l_run[8];
    #pragma unroll
    for (int r = 0; r < 8; ++r) { m_run[r] = -1e30f; l_run[r] = 0.0f; }
    const float scale = 0.125f;                  // 1/sqrt(64)
    const int qrow0 = q0 + w * 16 + lh * 8;      // + r = global query row

    const int nkb = (q0 + AQB) / AKB;            // causal coverage
    for (int kb = 0; kb < nkb; ++kb) {
        __syncthreads();                         // previous ks/vT reads done
        {   // stage k (natural) and v (transposed): one uint4 each per thread
            const int row = t >> 3, c = (t & 7) * 8;
            const size_t base = ((size_t)(b * SEQ + kb * AKB + row)) * DMODEL
                              + h * DHEAD + c;
            *(uint4*)&ks[row * KSTR + c] = *(const uint4*)(kB + base);
            uint4 vv = *(const uint4*)(vB + base);
            const unsigned short* pv = (const unsigned short*)&vv;
            #pragma unroll
            for (int j = 0; j < 8; ++j)
                vT[(c + j) * VSTR + row] = pv[j];
        }
        __syncthreads();

        // ---- scores: S = q @ k^T, two 16-key subtiles ----
        v8f sAcc[2] = {};
        #pragma unroll
        for (int sub = 0; sub < 2; ++sub) {
            #pragma unroll
            for (int f = 0; f < 2; ++f) {
                union { unsigned u[8]; v16bf v; } tk_;
                const int krow = sub * 16 + lmod;
                #pragma unroll
                for (int r = 0; r < 8; ++r)
                    tk_.u[r] = *(const unsigned*)&ks[krow * KSTR + f * 32
                                                    + ((r < 4) ? (lh * 8 + 2 * r) : (16 + lh * 8 + 2 * (r - 4)))];
                sAcc[sub] = __builtin_amdgcn_wmma_f32_16x16x32_bf16(
                    false, qfrag[f], false, tk_.v, (short)0, sAcc[sub],
                    false, false);
            }
        }

        // ---- scale + causal mask, extract ----
        float s0[8], s1[8];
        #pragma unroll
        for (int r = 0; r < 8; ++r) {
            const int qrow = qrow0 + r;
            const int c0 = kb * AKB + lmod;
            const int c1 = kb * AKB + 16 + lmod;
            s0[r] = (c0 > qrow) ? -1e30f : sAcc[0][r] * scale;
            s1[r] = (c1 > qrow) ? -1e30f : sAcc[1][r] * scale;
        }

        // ---- online softmax row stats ----
        float factor[8], p0[8], p1[8];
        #pragma unroll
        for (int r = 0; r < 8; ++r) {
            float m = fmaxf(s0[r], s1[r]);
            #pragma unroll
            for (int off = 1; off < 16; off <<= 1)
                m = fmaxf(m, __shfl_xor(m, off, 32));
            const float mn = fmaxf(m_run[r], m);
            factor[r] = __expf(m_run[r] - mn);
            m_run[r]  = mn;
            p0[r] = __expf(s0[r] - mn);
            p1[r] = __expf(s1[r] - mn);
            float sum = p0[r] + p1[r];
            #pragma unroll
            for (int off = 1; off < 16; off <<= 1)
                sum += __shfl_xor(sum, off, 32);
            l_run[r] = l_run[r] * factor[r] + sum;
        }
        #pragma unroll
        for (int di = 0; di < 4; ++di)
            #pragma unroll
            for (int r = 0; r < 8; ++r)
                accC[di][r] *= factor[r];

        // ---- repack P (C-layout) -> A-layout via per-wave LDS scratch ----
        #pragma unroll
        for (int r = 0; r < 8; ++r) {
            const int prow = r + lh * 8;
            ps[w][prow * PSTR + lmod]      = f2bf(p0[r]);
            ps[w][prow * PSTR + 16 + lmod] = f2bf(p1[r]);
        }
        asm volatile("s_wait_dscnt 0x0" ::: "memory");   // wave-local RAW

        union { unsigned u[8]; v16bf v; } tp_;
        #pragma unroll
        for (int r = 0; r < 8; ++r) {
            const int kk = (r < 4) ? (lh * 8 + 2 * r)
                                   : (16 + lh * 8 + 2 * (r - 4));
            tp_.u[r] = *(const unsigned*)&ps[w][lmod * PSTR + kk];
        }

        // ---- ctx += P @ V : 4 dh-subtiles ----
        #pragma unroll
        for (int di = 0; di < 4; ++di) {
            union { unsigned u[8]; v16bf v; } tv_;
            const int dhrow = di * 16 + lmod;
            #pragma unroll
            for (int r = 0; r < 8; ++r)
                tv_.u[r] = *(const unsigned*)&vT[dhrow * VSTR + ((r < 4) ? (lh * 8 + 2 * r) : (16 + lh * 8 + 2 * (r - 4)))];
            accC[di] = __builtin_amdgcn_wmma_f32_16x16x32_bf16(
                false, tp_.v, false, tv_.v, (short)0, accC[di], false, false);
        }
    }

    // ---- normalize and store ctx (bf16) ----
    #pragma unroll
    for (int r = 0; r < 8; ++r) {
        const float inv = 1.0f / l_run[r];
        const int row = qrow0 + r;
        #pragma unroll
        for (int di = 0; di < 4; ++di)
            ctxB[((size_t)(b * SEQ + row)) * DMODEL + h * DHEAD + di * 16 + lmod]
                = f2bf(accC[di][r] * inv);
    }
}

// ---------------------------------------------------------------------------
// (x + res) -> LayerNorm (jnp.var semantics), per-row. res nullable.
// ---------------------------------------------------------------------------
__global__ __launch_bounds__(256)
void ln_res_kernel(const float* __restrict__ x, const float* res,
                   const float* __restrict__ g, const float* __restrict__ bia,
                   float* __restrict__ out)
{
    __shared__ float ssum[256], ssq[256];
    const int row = blockIdx.x;
    float s = 0.0f, sq = 0.0f;
    for (int c = threadIdx.x; c < DMODEL; c += 256) {
        float vv = x[(size_t)row * DMODEL + c];
        if (res) vv += res[(size_t)row * DMODEL + c];
        s += vv; sq += vv * vv;
    }
    ssum[threadIdx.x] = s; ssq[threadIdx.x] = sq;
    __syncthreads();
    for (int st = 128; st > 0; st >>= 1) {
        if (threadIdx.x < st) {
            ssum[threadIdx.x] += ssum[threadIdx.x + st];
            ssq[threadIdx.x]  += ssq[threadIdx.x + st];
        }
        __syncthreads();
    }
    const float mu  = ssum[0] * (1.0f / DMODEL);
    const float var = ssq[0] * (1.0f / DMODEL) - mu * mu;
    const float inv = rsqrtf(var + 1e-5f);
    for (int c = threadIdx.x; c < DMODEL; c += 256) {
        float vv = x[(size_t)row * DMODEL + c];
        if (res) vv += res[(size_t)row * DMODEL + c];
        out[(size_t)row * DMODEL + c] = (vv - mu) * inv * g[c] + bia[c];
    }
}

// ---------------------------------------------------------------------------
// Host launcher
// ---------------------------------------------------------------------------
extern "C" void kernel_launch(void* const* d_in, const int* in_sizes, int n_in,
                              void* d_out, int out_size, void* d_ws, size_t ws_size,
                              hipStream_t stream)
{
    (void)in_sizes; (void)n_in; (void)out_size; (void)ws_size;
    const int*   x     = (const int*)  d_in[0];
    /* d_in[1] = mask (unused; causal handled analytically) */
    const float* embed = (const float*)d_in[2];
    const float* wq = (const float*)d_in[3];  const float* bq = (const float*)d_in[4];
    const float* wk = (const float*)d_in[5];  const float* bk = (const float*)d_in[6];
    const float* wv = (const float*)d_in[7];  const float* bv = (const float*)d_in[8];
    const float* wo = (const float*)d_in[9];  const float* bo = (const float*)d_in[10];
    const float* w1 = (const float*)d_in[11]; const float* b1 = (const float*)d_in[12];
    const float* w2 = (const float*)d_in[13]; const float* b2 = (const float*)d_in[14];
    const float* ln1g = (const float*)d_in[15]; const float* ln1b = (const float*)d_in[16];
    const float* ln2g = (const float*)d_in[17]; const float* ln2b = (const float*)d_in[18];
    const float* lnfg = (const float*)d_in[19]; const float* lnfb = (const float*)d_in[20];
    const float* pw = (const float*)d_in[21]; const float* pb = (const float*)d_in[22];
    float* out = (float*)d_out;

    char* ws = (char*)d_ws;
    size_t off = 0;
    auto take = [&](size_t bytes) -> void* {
        void* p = (void*)(ws + off);
        off += (bytes + 255) & ~(size_t)255;
        return p;
    };

    const size_t DD = (size_t)DMODEL * DMODEL;
    const size_t DF = (size_t)DMODEL * FFN;
    unsigned short* wqT = (unsigned short*)take(NLAYER * DD * 2);
    unsigned short* wkT = (unsigned short*)take(NLAYER * DD * 2);
    unsigned short* wvT = (unsigned short*)take(NLAYER * DD * 2);
    unsigned short* woT = (unsigned short*)take(NLAYER * DD * 2);
    unsigned short* w1T = (unsigned short*)take(NLAYER * DF * 2);
    unsigned short* w2T = (unsigned short*)take(NLAYER * DF * 2);
    unsigned short* pwT = (unsigned short*)take((size_t)VOCAB * DMODEL * 2);
    float*          h    = (float*)take((size_t)MTOK * DMODEL * 4);
    unsigned short* hb   = (unsigned short*)take((size_t)MTOK * DMODEL * 2);
    unsigned short* qBuf = (unsigned short*)take((size_t)MTOK * DMODEL * 2);
    unsigned short* kBuf = (unsigned short*)take((size_t)MTOK * DMODEL * 2);
    unsigned short* vBuf = (unsigned short*)take((size_t)MTOK * DMODEL * 2);
    unsigned short* ctxB = (unsigned short*)take((size_t)MTOK * DMODEL * 2);
    unsigned short* f1B  = (unsigned short*)take((size_t)MTOK * FFN * 2);
    float*          tf32 = (float*)take((size_t)MTOK * DMODEL * 4);

    const dim3 tb(32, 8);

    // ---- one-time weight transpose+convert to bf16 [N][K] ----
    for (int l = 0; l < NLAYER; ++l) {
        transpose_conv_kernel<<<dim3(DMODEL/32, DMODEL/32), tb, 0, stream>>>(
            wq + l * DD, wqT + l * DD, DMODEL, DMODEL);
        transpose_conv_kernel<<<dim3(DMODEL/32, DMODEL/32), tb, 0, stream>>>(
            wk + l * DD, wkT + l * DD, DMODEL, DMODEL);
        transpose_conv_kernel<<<dim3(DMODEL/32, DMODEL/32), tb, 0, stream>>>(
            wv + l * DD, wvT + l * DD, DMODEL, DMODEL);
        transpose_conv_kernel<<<dim3(DMODEL/32, DMODEL/32), tb, 0, stream>>>(
            wo + l * DD, woT + l * DD, DMODEL, DMODEL);
        transpose_conv_kernel<<<dim3(FFN/32, DMODEL/32), tb, 0, stream>>>(
            w1 + l * DF, w1T + l * DF, DMODEL, FFN);       // out: [FFN][DMODEL]
        transpose_conv_kernel<<<dim3(DMODEL/32, FFN/32), tb, 0, stream>>>(
            w2 + l * DF, w2T + l * DF, FFN, DMODEL);       // out: [DMODEL][FFN]
    }
    transpose_conv_kernel<<<dim3(VOCAB/32, DMODEL/32), tb, 0, stream>>>(
        pw, pwT, DMODEL, VOCAB);                           // out: [VOCAB][DMODEL]

    // ---- embedding + PE ----
    embed_pe_kernel<<<MTOK, 256, 0, stream>>>(x, embed, h);

    const int nMD = MTOK * DMODEL;
    const dim3 gMD((nMD + 255) / 256);
    const dim3 gemmD(DMODEL / BN, MTOK / BM);   // N=1024
    const dim3 gemmF(FFN / BN,    MTOK / BM);   // N=4096
    const dim3 gemmV(VOCAB / BN,  MTOK / BM);   // N=32000
    const dim3 attnG(SEQ / AQB, BATCH * NHEAD);

    for (int l = 0; l < NLAYER; ++l) {
        conv_bf16_kernel<<<gMD, 256, 0, stream>>>(h, hb, nMD);
        // q,k,v projections -> bf16 directly (mode 2)
        gemm_bf16_kernel<<<gemmD, 256, 0, stream>>>(hb, wqT + l * DD,
            bq + l * DMODEL, qBuf, MTOK, DMODEL, DMODEL, 2);
        gemm_bf16_kernel<<<gemmD, 256, 0, stream>>>(hb, wkT + l * DD,
            bk + l * DMODEL, kBuf, MTOK, DMODEL, DMODEL, 2);
        gemm_bf16_kernel<<<gemmD, 256, 0, stream>>>(hb, wvT + l * DD,
            bv + l * DMODEL, vBuf, MTOK, DMODEL, DMODEL, 2);
        // WMMA flash attention -> bf16 ctx
        attn_wmma_kernel<<<attnG, 256, 0, stream>>>(qBuf, kBuf, vBuf, ctxB);
        // output projection -> f32
        gemm_bf16_kernel<<<gemmD, 256, 0, stream>>>(ctxB, woT + l * DD,
            bo + l * DMODEL, tf32, MTOK, DMODEL, DMODEL, 0);
        ln_res_kernel<<<MTOK, 256, 0, stream>>>(h, tf32,
            ln1g + l * DMODEL, ln1b + l * DMODEL, h);
        // FFN: W1 + bias + ReLU -> bf16 (mode 3), then W2 -> f32
        conv_bf16_kernel<<<gMD, 256, 0, stream>>>(h, hb, nMD);
        gemm_bf16_kernel<<<gemmF, 256, 0, stream>>>(hb, w1T + l * DF,
            b1 + l * FFN, f1B, MTOK, FFN, DMODEL, 3);
        gemm_bf16_kernel<<<gemmD, 256, 0, stream>>>(f1B, w2T + l * DF,
            b2 + l * DMODEL, tf32, MTOK, DMODEL, FFN, 0);
        ln_res_kernel<<<MTOK, 256, 0, stream>>>(h, tf32,
            ln2g + l * DMODEL, ln2b + l * DMODEL, h);
    }

    // final LN + vocab projection
    ln_res_kernel<<<MTOK, 256, 0, stream>>>(h, (const float*)nullptr, lnfg, lnfb, h);
    conv_bf16_kernel<<<gMD, 256, 0, stream>>>(h, hb, nMD);
    gemm_bf16_kernel<<<gemmV, 256, 0, stream>>>(hb, pwT, pb, out,
        MTOK, VOCAB, DMODEL, 0);
}
